// Multihead_89962384982768
// MI455X (gfx1250) — matrix operand order repaired and checked by reading, not compiled, not run on hardware
//
#include <hip/hip_runtime.h>
#include <hip/hip_bf16.h>

// ---------------------------------------------------------------------------
// Causal multi-head attention for MI455X (gfx1250), wave32 + WMMA f16 16x16x32.
//
// Roofline: ~52 GFLOP total, ~100MB HBM traffic (everything else lives in the
// 192MB L2) -> memory-bound; fp16 operands + f32 WMMA accumulation is the
// right precision/data-movement point. Attention kernel computes S^T and O^T
// so that softmax is lane-local (+one shfl_xor(16)) and the P operand for the
// PV matmul is rebuilt in-register from WMMA accumulators (no LDS transpose).
// ---------------------------------------------------------------------------

typedef __attribute__((ext_vector_type(16))) _Float16 v16h;
typedef __attribute__((ext_vector_type(8)))  _Float16 v8h;
typedef __attribute__((ext_vector_type(8)))  float    v8f;

#define EMB   384
#define SEQ   256
#define NB    128
#define NH    6
#define HD    64
#define MTOT  (NB * SEQ)   // 32768 rows

// ---- fragment loaders (this code's contiguous k order, same for A and B) ----------------------
// A-matrix 16x32 f16: lane m = L&15; K base c0 = (L>=16 ? 16 : 0):
//   elements 0..15 = K c0..c0+15.
__device__ inline v16h load_fragA(const _Float16* p, int row, int k, int ld) {
  const int lane = threadIdx.x & 31;
  const _Float16* q = p + (size_t)(row + (lane & 15)) * ld + k + ((lane >> 4) << 4);
  v8h lo = *(const v8h*)(q);
  v8h hi = *(const v8h*)(q + 8);
  v16h f;
#pragma unroll
  for (int i = 0; i < 8; ++i) { f[i] = lo[i]; f[i + 8] = hi[i]; }
  return f;
}

// B-matrix 32x16 f16 loaded from row-major [n][k] storage:
// lane n = L&15; lanes 0-15 hold K=0..15, lanes 16-31 hold K=16..31.
__device__ inline v16h load_fragB(const _Float16* p, int ncol, int k, int ld) {
  const int lane = threadIdx.x & 31;
  const _Float16* q = p + (size_t)(ncol + (lane & 15)) * ld + k + ((lane >> 4) << 4);
  v8h lo = *(const v8h*)(q);
  v8h hi = *(const v8h*)(q + 8);
  v16h f;
#pragma unroll
  for (int i = 0; i < 8; ++i) { f[i] = lo[i]; f[i + 8] = hi[i]; }
  return f;
}

__device__ inline v8f wmma16(v16h a, v16h b, v8f c) {
  return __builtin_amdgcn_wmma_f32_16x16x32_f16(false, a, false, b, (short)0, c,
                                                false, false);
}

__device__ inline v8f vzero8() {
  v8f z;
#pragma unroll
  for (int i = 0; i < 8; ++i) z[i] = 0.0f;
  return z;
}

// ---- prep kernels ----------------------------------------------------------
__global__ void cvt_f16_kernel(const float* __restrict__ src,
                               _Float16* __restrict__ dst, int n) {
  int i = blockIdx.x * blockDim.x + threadIdx.x;
  const int stride = gridDim.x * blockDim.x;
  for (; i < n; i += stride) dst[i] = (_Float16)src[i];
}

// Wt[n*EMB + k] = W[k*EMB + n]  (weights are tiny; stays in L2 forever)
__global__ void transpose_w_kernel(const float* __restrict__ W,
                                   _Float16* __restrict__ Wt) {
  int i = blockIdx.x * blockDim.x + threadIdx.x;
  if (i < EMB * EMB) {
    int n = i / EMB, k = i % EMB;
    Wt[i] = (_Float16)W[k * EMB + n];
  }
}

// ---- fused QKV projection --------------------------------------------------
// Each wave computes one 32x64 output tile of one of {Q,K,V}.
// Q,K stored per-head row-major [b,h,t,d]; V stored transposed [b,h,d,t].
__global__ void qkv_kernel(const _Float16* __restrict__ xh,
                           const _Float16* __restrict__ wtq,
                           const _Float16* __restrict__ wtk,
                           const _Float16* __restrict__ wtv,
                           const float* __restrict__ bq,
                           const float* __restrict__ bk,
                           const float* __restrict__ bv,
                           _Float16* __restrict__ Qh,
                           _Float16* __restrict__ Kh,
                           _Float16* __restrict__ Vt) {
  const int lane = threadIdx.x & 31;
  const int wave = threadIdx.x >> 5;
  const int tile = blockIdx.x * 8 + wave;          // 3 * 1024 * 6 tiles total
  const int mat  = tile / (1024 * 6);
  const int rem  = tile % (1024 * 6);
  const int mrow = (rem / 6) * 32;
  const int ncol = (rem % 6) * 64;

  const _Float16* Wt   = (mat == 0) ? wtq : (mat == 1) ? wtk : wtv;
  const float*    bias = (mat == 0) ? bq  : (mat == 1) ? bk  : bv;

  v8f c[2][4];
#pragma unroll
  for (int mt = 0; mt < 2; ++mt)
#pragma unroll
    for (int nt = 0; nt < 4; ++nt) c[mt][nt] = vzero8();

  for (int ks = 0; ks < 12; ++ks) {
    v16h a0 = load_fragA(xh, mrow,      ks * 32, EMB);
    v16h a1 = load_fragA(xh, mrow + 16, ks * 32, EMB);
#pragma unroll
    for (int nt = 0; nt < 4; ++nt) {
      v16h b = load_fragB(Wt, ncol + nt * 16, ks * 32, EMB);
      c[0][nt] = wmma16(a0, b, c[0][nt]);
      c[1][nt] = wmma16(a1, b, c[1][nt]);
    }
  }

  const int hi8 = (lane >> 4) << 3;
#pragma unroll
  for (int nt = 0; nt < 4; ++nt) {
    const int g  = ncol + nt * 16 + (lane & 15);   // global output column
    const int hh = g >> 6;
    const int dd = g & 63;
    const float bval = bias[g];
#pragma unroll
    for (int mt = 0; mt < 2; ++mt) {
      const int m0 = mrow + mt * 16 + hi8;         // first global row (t index)
      const int bb = m0 >> 8;
      const int t0 = m0 & 255;
      const size_t head = (size_t)(bb * NH + hh);
      if (mat == 2) {
        // V transposed: rows t are contiguous per lane -> packed 16B store
        v8h pk;
#pragma unroll
        for (int v = 0; v < 8; ++v) pk[v] = (_Float16)(c[mt][nt][v] + bval);
        *(v8h*)(Vt + (head * HD + dd) * SEQ + t0) = pk;
      } else {
        _Float16* dst = ((mat == 0) ? Qh : Kh) + (head * SEQ + t0) * HD + dd;
#pragma unroll
        for (int v = 0; v < 8; ++v)
          dst[(size_t)v * HD] = (_Float16)(c[mt][nt][v] + bval);
      }
    }
  }
}

// ---- flash attention (causal) ----------------------------------------------
// One workgroup per (b,h); wave w owns queries [32w, 32w+32).
// Computes S^T = K*Q^T and O^T = V^T*P^T so P never leaves registers.
__global__ void attn_kernel(const _Float16* __restrict__ Qh,
                            const _Float16* __restrict__ Kh,
                            const _Float16* __restrict__ Vt,
                            _Float16* __restrict__ Oh) {
  const int lane  = threadIdx.x & 31;
  const int wave  = threadIdx.x >> 5;
  const int bh    = blockIdx.x;
  const int bb    = bh / NH;
  const int hh    = bh % NH;
  const int qbase = wave * 32;
  const int hi8   = (lane >> 4) << 3;
  const bool lolane = (lane < 16);

  const _Float16* Qp = Qh + (size_t)bh * SEQ * HD;
  const _Float16* Kp = Kh + (size_t)bh * SEQ * HD;
  const _Float16* Vp = Vt + (size_t)bh * HD * SEQ;

  // Q as B-operand (lane = query column), loaded once.
  v16h qf[2][2];
#pragma unroll
  for (int nt = 0; nt < 2; ++nt)
#pragma unroll
    for (int kk = 0; kk < 2; ++kk)
      qf[nt][kk] = load_fragB(Qp, qbase + nt * 16, kk * 32, HD);

  v8f o[4][2];
#pragma unroll
  for (int dt = 0; dt < 4; ++dt)
#pragma unroll
    for (int nt = 0; nt < 2; ++nt) o[dt][nt] = vzero8();

  float mrun[2] = {-1e30f, -1e30f};
  float lrun[2] = {0.0f, 0.0f};

  const int nblocks = (qbase + 95) >> 6;           // causal: only needed key blocks
  for (int blk = 0; blk < nblocks; ++blk) {
    const int kb = blk * 64;

    // S^T tiles [key 64][query 32]
    v8f s[4][2];
#pragma unroll
    for (int mt = 0; mt < 4; ++mt)
#pragma unroll
      for (int nt = 0; nt < 2; ++nt) s[mt][nt] = vzero8();
#pragma unroll
    for (int kk = 0; kk < 2; ++kk)
#pragma unroll
      for (int mt = 0; mt < 4; ++mt) {
        v16h kf = load_fragA(Kp, kb + mt * 16, kk * 32, HD);
#pragma unroll
        for (int nt = 0; nt < 2; ++nt)
          s[mt][nt] = wmma16(kf, qf[nt][kk], s[mt][nt]);
      }

    // scale + causal mask + online softmax (lane owns one query per nt)
#pragma unroll
    for (int nt = 0; nt < 2; ++nt) {
      const int q = qbase + nt * 16 + (lane & 15);
      float rm = -1e30f;
#pragma unroll
      for (int mt = 0; mt < 4; ++mt)
#pragma unroll
        for (int v = 0; v < 8; ++v) {
          const int key = kb + mt * 16 + hi8 + v;
          float val = s[mt][nt][v] * 0.125f;       // 1/sqrt(64)
          val = (key > q) ? -1e30f : val;
          s[mt][nt][v] = val;
          rm = fmaxf(rm, val);
        }
      rm = fmaxf(rm, __shfl_xor(rm, 16, 32));      // keys split across lane halves
      const float mnew  = fmaxf(mrun[nt], rm);
      const float alpha = __expf(mrun[nt] - mnew);
      float rs = 0.0f;
#pragma unroll
      for (int mt = 0; mt < 4; ++mt)
#pragma unroll
        for (int v = 0; v < 8; ++v) {
          const float p = __expf(s[mt][nt][v] - mnew);
          s[mt][nt][v] = p;
          rs += p;
        }
      rs += __shfl_xor(rs, 16, 32);
      lrun[nt] = lrun[nt] * alpha + rs;
      mrun[nt] = mnew;
#pragma unroll
      for (int dt = 0; dt < 4; ++dt)
#pragma unroll
        for (int v = 0; v < 8; ++v) o[dt][nt][v] *= alpha;
    }

    // Build P^T B-fragments from the S^T accumulators and do O^T += V^T * P^T.
    v16h pf[2][2];
#pragma unroll
    for (int nt = 0; nt < 2; ++nt)
#pragma unroll
      for (int kk = 0; kk < 2; ++kk) {
        v16h f;
#pragma unroll
        for (int j = 0; j < 8; ++j) {
          const float own0 = s[2 * kk][nt][j];       // keys kb+32kk + (0..7 | 8..15)
          const float own1 = s[2 * kk + 1][nt][j];   // keys kb+32kk + (16..23 | 24..31)
          const float oth0 = __shfl_xor(own0, 16, 32);
          const float oth1 = __shfl_xor(own1, 16, 32);
          f[j]     = (_Float16)(lolane ? own0 : oth1);
          f[j + 8] = (_Float16)(lolane ? oth0 : own1);
        }
        pf[nt][kk] = f;
      }
#pragma unroll
    for (int kk = 0; kk < 2; ++kk)
#pragma unroll
      for (int dt = 0; dt < 4; ++dt) {
        v16h vf = load_fragA(Vp, dt * 16, kb + kk * 32, SEQ);
#pragma unroll
        for (int nt = 0; nt < 2; ++nt)
          o[dt][nt] = wmma16(vf, pf[nt][kk], o[dt][nt]);
      }
  }

  // O^T: lane = query column, d contiguous per lane -> packed 16B stores.
#pragma unroll
  for (int nt = 0; nt < 2; ++nt) {
    const int q = qbase + nt * 16 + (lane & 15);
    const float inv = 1.0f / lrun[nt];
#pragma unroll
    for (int dt = 0; dt < 4; ++dt) {
      v8h pk;
#pragma unroll
      for (int v = 0; v < 8; ++v) pk[v] = (_Float16)(o[dt][nt][v] * inv);
      *(v8h*)(Oh + ((size_t)(bb * SEQ + q)) * EMB + hh * 64 + dt * 16 + hi8) = pk;
    }
  }
}

// ---- output projection (fp32 result) ---------------------------------------
__global__ void proj_kernel(const _Float16* __restrict__ Oh,
                            const _Float16* __restrict__ wtp,
                            const float* __restrict__ bp,
                            float* __restrict__ out) {
  const int lane = threadIdx.x & 31;
  const int wave = threadIdx.x >> 5;
  const int tile = blockIdx.x * 8 + wave;          // 1024 * 6 tiles
  const int mrow = (tile / 6) * 32;
  const int ncol = (tile % 6) * 64;

  v8f c[2][4];
#pragma unroll
  for (int mt = 0; mt < 2; ++mt)
#pragma unroll
    for (int nt = 0; nt < 4; ++nt) c[mt][nt] = vzero8();

  for (int ks = 0; ks < 12; ++ks) {
    v16h a0 = load_fragA(Oh, mrow,      ks * 32, EMB);
    v16h a1 = load_fragA(Oh, mrow + 16, ks * 32, EMB);
#pragma unroll
    for (int nt = 0; nt < 4; ++nt) {
      v16h b = load_fragB(wtp, ncol + nt * 16, ks * 32, EMB);
      c[0][nt] = wmma16(a0, b, c[0][nt]);
      c[1][nt] = wmma16(a1, b, c[1][nt]);
    }
  }

  const int hi8 = (lane >> 4) << 3;
#pragma unroll
  for (int nt = 0; nt < 4; ++nt) {
    const int g = ncol + nt * 16 + (lane & 15);
    const float bval = bp[g];
#pragma unroll
    for (int mt = 0; mt < 2; ++mt) {
      const int m0 = mrow + mt * 16 + hi8;
#pragma unroll
      for (int v = 0; v < 8; ++v)
        out[(size_t)(m0 + v) * EMB + g] = c[mt][nt][v] + bval;
    }
  }
}

// ---------------------------------------------------------------------------
extern "C" void kernel_launch(void* const* d_in, const int* in_sizes, int n_in,
                              void* d_out, int out_size, void* d_ws, size_t ws_size,
                              hipStream_t stream) {
  (void)in_sizes; (void)n_in; (void)out_size; (void)ws_size;
  const float* x  = (const float*)d_in[0];
  const float* Wq = (const float*)d_in[1];
  const float* bq = (const float*)d_in[2];
  const float* Wk = (const float*)d_in[3];
  const float* bk = (const float*)d_in[4];
  const float* Wv = (const float*)d_in[5];
  const float* bv = (const float*)d_in[6];
  const float* Wp = (const float*)d_in[7];
  const float* bp = (const float*)d_in[8];
  float* out = (float*)d_out;

  const size_t nx = (size_t)MTOT * EMB;            // 12,582,912 elements
  _Float16* xh  = (_Float16*)d_ws;                 // also reused as Oh later
  _Float16* wtq = xh + nx;
  _Float16* wtk = wtq + EMB * EMB;
  _Float16* wtv = wtk + EMB * EMB;
  _Float16* wtp = wtv + EMB * EMB;
  _Float16* Qh  = wtp + EMB * EMB;
  _Float16* Kh  = Qh + nx;
  _Float16* Vt  = Kh + nx;
  _Float16* Oh  = xh;                              // xh is dead after qkv_kernel

  cvt_f16_kernel<<<2048, 256, 0, stream>>>(x, xh, (int)nx);
  transpose_w_kernel<<<576, 256, 0, stream>>>(Wq, wtq);
  transpose_w_kernel<<<576, 256, 0, stream>>>(Wk, wtk);
  transpose_w_kernel<<<576, 256, 0, stream>>>(Wv, wtv);
  transpose_w_kernel<<<576, 256, 0, stream>>>(Wp, wtp);

  qkv_kernel<<<2304, 256, 0, stream>>>(xh, wtq, wtk, wtv, bq, bk, bv, Qh, Kh, Vt);
  attn_kernel<<<NB * NH, 256, 0, stream>>>(Qh, Kh, Vt, Oh);
  proj_kernel<<<768, 256, 0, stream>>>(Oh, wtp, bp, out);
}
